// EqModelComplex_38027640439012
// MI455X (gfx1250) — matrix operand order repaired and checked by reading, not compiled, not run on hardware
//
#include <hip/hip_runtime.h>
#include <hip/hip_bf16.h>
#include <math.h>

typedef __bf16 bf16_t;
typedef __attribute__((ext_vector_type(16))) __bf16 v16bf;
typedef __attribute__((ext_vector_type(8)))  float  v8f;
typedef __attribute__((ext_vector_type(4)))  int    v4i;

#define DEV_INLINE __device__ __forceinline__

constexpr int Bc = 2, Lc = 1024, Dc = 1024, Hc = 16, HDc = 64, HIDc = 4096;
constexpr int MR = Bc * Lc;            // 2048 rows
constexpr float SCALEc = 0.125f;       // HD^-0.5
constexpr float EPSc = 1e-6f;
constexpr size_t NB   = (size_t)Bc * Lc * Dc;     // 2M
constexpr size_t NHID = (size_t)Bc * Lc * HIDc;   // 8M

// ---------------- WMMA fragment helpers (bf16 16x16x32, wave32) ----------------
// Generic two-halves loader: A and B fragments both use gap=16 elements.
DEV_INLINE v16bf load_two(const bf16_t* p, int gap) {
    union { v4i q[2]; v16bf v; } u;
    u.q[0] = *(const v4i*)(p);
    u.q[1] = *(const v4i*)(p + gap);
    return u.v;
}
// A-matrix 16x32 (MxK): lanes 0-15 row M=lane, K=0..7 & 16..23; lanes 16-31 K=8..15 & 24..31.
DEV_INLINE v16bf load_frag_a(const bf16_t* base, int ld, int lane) {
    int row = lane & 15;
    int kh  = (lane >> 4) & 1;
    return load_two(base + (size_t)row * ld + kh * 8, 16);
}
// B-matrix 32x16 (KxN): lanes 0-15 col n=lane, K=0..7 & 16..23; lanes 16-31 col n-16, K=8..15 & 24..31.
// `base` points at a [16 x K]-row-major tile where row = output column (torch W[out,in]).
DEV_INLINE v16bf load_frag_b(const bf16_t* base, int ld, int lane) {
    int col = lane & 15;
    int kh  = (lane >> 4) & 1;
    return load_two(base + (size_t)col * ld + kh * 8, 16);
}
DEV_INLINE v8f wmma_bf16(v16bf a, v16bf b, v8f c) {
    return __builtin_amdgcn_wmma_f32_16x16x32_bf16(false, a, false, b, (short)0, c, false, false);
}

// ---------------- fp32 -> bf16 convert ----------------
__global__ void f2b_kernel(const float* __restrict__ in, bf16_t* __restrict__ out, int n) {
    int t = blockIdx.x * blockDim.x + threadIdx.x;
    if (t < n) out[t] = (bf16_t)in[t];
}

// ---------------- complex LayerNorm -> bf16 ----------------
__global__ void cln_kernel(const float* __restrict__ xr, const float* __restrict__ xi,
                           const float* __restrict__ gr, const float* __restrict__ gi,
                           const float* __restrict__ bbr, const float* __restrict__ bbi,
                           bf16_t* __restrict__ outr, bf16_t* __restrict__ outi) {
    __shared__ float red[256];
    int row = blockIdx.x;                          // B*L rows
    const float* pr = xr + (size_t)row * Dc;
    const float* pi = xi + (size_t)row * Dc;
    float lr[4], li[4];
    float sr = 0.f, si = 0.f;
#pragma unroll
    for (int k = 0; k < 4; ++k) {
        int d = threadIdx.x + k * 256;
        lr[k] = pr[d]; li[k] = pi[d];
        sr += lr[k]; si += li[k];
    }
    auto block_sum = [&](float v) -> float {
        red[threadIdx.x] = v; __syncthreads();
        for (int st = 128; st > 0; st >>= 1) {
            if (threadIdx.x < (unsigned)st) red[threadIdx.x] += red[threadIdx.x + st];
            __syncthreads();
        }
        float r = red[0]; __syncthreads();
        return r;
    };
    float mr = block_sum(sr) * (1.0f / Dc);
    float mi = block_sum(si) * (1.0f / Dc);
    float sv = 0.f;
#pragma unroll
    for (int k = 0; k < 4; ++k) {
        float cr = lr[k] - mr, ci = li[k] - mi;
        sv += cr * cr + ci * ci;
    }
    float var = block_sum(sv) * (1.0f / Dc);
    float inv = 1.0f / sqrtf(var + EPSc);
    bf16_t* qr = outr + (size_t)row * Dc;
    bf16_t* qi = outi + (size_t)row * Dc;
#pragma unroll
    for (int k = 0; k < 4; ++k) {
        int d = threadIdx.x + k * 256;
        float nr = (lr[k] - mr) * inv, ni = (li[k] - mi) * inv;
        float g_r = gr[d], g_i = gi[d];
        qr[d] = (bf16_t)(nr * g_r - ni * g_i + bbr[d]);
        qi[d] = (bf16_t)(nr * g_i + ni * g_r + bbi[d]);
    }
}

// ---------------- complex GEMM via bf16 WMMA, 32x32 register-blocked wave tile ----
// Y = X @ W^T complex; X [M,K] row-major bf16, W [N,K] row-major bf16 (torch layout).
// Each wave: 2 M-tiles x 2 N-tiles => 16 WMMAs per K=32 step on 8 fragment loads.
// ~190 VGPRs: fits without spilling under launch_bounds(256,1).
// mode 0: fp32 store, +bias (if non-null), +residual (if non-null)
// mode 1: +bias, ModReLU, bf16 store (for FFN hidden)
__global__ __launch_bounds__(256, 1)
void cgemm_kernel(const bf16_t* __restrict__ Xr, const bf16_t* __restrict__ Xi,
                  const bf16_t* __restrict__ Wr, const bf16_t* __restrict__ Wi,
                  float* Yr, float* Yi, bf16_t* Ybr, bf16_t* Ybi,
                  const float* biasr, const float* biasi,
                  const float* Rr, const float* Ri,
                  const float* modb,
                  int M, int N, int K, int mode) {
    int lane = threadIdx.x & 31;
    int wid  = blockIdx.x * (blockDim.x >> 5) + (threadIdx.x >> 5);
    int nblk = N >> 5;                       // 32-wide N blocks
    int mb_ = wid / nblk, nb_ = wid % nblk;
    int m0 = mb_ * 32, n0 = nb_ * 32;

    // hoisted per-lane addressing
    int row = lane & 15;
    int kh  = (lane >> 4) & 1;
    size_t aoff = (size_t)row * K + kh * 8;
    size_t boff = (size_t)row * K + kh * 8;
    const bf16_t* par[2]; const bf16_t* pai[2];
    const bf16_t* pbr[2]; const bf16_t* pbi[2];
#pragma unroll
    for (int mi = 0; mi < 2; ++mi) {
        par[mi] = Xr + (size_t)(m0 + 16 * mi) * K + aoff;
        pai[mi] = Xi + (size_t)(m0 + 16 * mi) * K + aoff;
    }
#pragma unroll
    for (int ni = 0; ni < 2; ++ni) {
        pbr[ni] = Wr + (size_t)(n0 + 16 * ni) * K + boff;
        pbi[ni] = Wi + (size_t)(n0 + 16 * ni) * K + boff;
    }

    v8f zero8 = {0,0,0,0,0,0,0,0};
    v8f acc_rr[2][2], acc_ii[2][2], acc_i[2][2];
#pragma unroll
    for (int mi = 0; mi < 2; ++mi)
#pragma unroll
        for (int ni = 0; ni < 2; ++ni) {
            acc_rr[mi][ni] = zero8; acc_ii[mi][ni] = zero8; acc_i[mi][ni] = zero8;
        }

    for (int k = 0; k < K; k += 32) {
        v16bf axr[2], axi[2], bwr[2], bwi[2];
#pragma unroll
        for (int mi = 0; mi < 2; ++mi) {
            axr[mi] = load_two(par[mi] + k, 16);
            axi[mi] = load_two(pai[mi] + k, 16);
        }
#pragma unroll
        for (int ni = 0; ni < 2; ++ni) {
            bwr[ni] = load_two(pbr[ni] + k, 16);
            bwi[ni] = load_two(pbi[ni] + k, 16);
        }
#pragma unroll
        for (int mi = 0; mi < 2; ++mi)
#pragma unroll
            for (int ni = 0; ni < 2; ++ni) {
                acc_rr[mi][ni] = wmma_bf16(axr[mi], bwr[ni], acc_rr[mi][ni]);
                acc_ii[mi][ni] = wmma_bf16(axi[mi], bwi[ni], acc_ii[mi][ni]);
                acc_i [mi][ni] = wmma_bf16(axr[mi], bwi[ni], acc_i [mi][ni]);
                acc_i [mi][ni] = wmma_bf16(axi[mi], bwr[ni], acc_i [mi][ni]);
            }
    }

    int lcol = lane & 15;
    int rofs = (lane >> 4) ? 8 : 0;
#pragma unroll
    for (int mi = 0; mi < 2; ++mi)
#pragma unroll
        for (int ni = 0; ni < 2; ++ni) {
            int col   = n0 + ni * 16 + lcol;
            int rbase = m0 + mi * 16 + rofs;
            float b_r = biasr ? biasr[col] : 0.f;
            float b_i = biasi ? biasi[col] : 0.f;
            float mb  = modb  ? modb[col]  : 0.f;
#pragma unroll
            for (int j = 0; j < 8; ++j) {
                size_t idx = (size_t)(rbase + j) * N + col;
                float yr = acc_rr[mi][ni][j] - acc_ii[mi][ni][j] + b_r;
                float yi = acc_i[mi][ni][j] + b_i;
                if (mode == 0) {
                    if (Rr) { yr += Rr[idx]; yi += Ri[idx]; }
                    Yr[idx] = yr; Yi[idx] = yi;
                } else {
                    float mag  = sqrtf(yr * yr + yi * yi);
                    float safe = mag > 0.f ? mag : 1.f;
                    float act  = fmaxf(mag + mb, 0.f);
                    float s    = act / safe;
                    Ybr[idx] = (bf16_t)(yr * s);
                    Ybi[idx] = (bf16_t)(yi * s);
                }
            }
        }
}

// ---------------- RoPE (fp32 in -> bf16 out) on Q and K ----------------
__global__ void rope_kernel(const float* __restrict__ Qr, const float* __restrict__ Qi,
                            const float* __restrict__ Kr, const float* __restrict__ Ki,
                            bf16_t* __restrict__ qbr, bf16_t* __restrict__ qbi,
                            bf16_t* __restrict__ kbr, bf16_t* __restrict__ kbi) {
    int t = blockIdx.x * blockDim.x + threadIdx.x;   // B*L*H*32
    if (t >= Bc * Lc * Hc * 32) return;
    int p = t & 31;
    int h = (t >> 5) & (Hc - 1);
    int l = (t >> 9) & (Lc - 1);
    int b = t >> 19;
    size_t base = ((size_t)(b * Lc + l)) * Dc + h * HDc;
    // inv_freq = 10000^(-2p/64); ln(10000) = 9.210340371976184
    float f = (float)l * expf(-(float)(2 * p) * (9.210340372f / 64.f));
    float c = cosf(f), s = sinf(f);
    size_t i0 = base + p, i1 = base + p + 32;
    float a0, a1;
    a0 = Qr[i0]; a1 = Qr[i1];
    qbr[i0] = (bf16_t)(a0 * c - a1 * s);  qbr[i1] = (bf16_t)(a1 * c + a0 * s);
    a0 = Qi[i0]; a1 = Qi[i1];
    qbi[i0] = (bf16_t)(a0 * c - a1 * s);  qbi[i1] = (bf16_t)(a1 * c + a0 * s);
    a0 = Kr[i0]; a1 = Kr[i1];
    kbr[i0] = (bf16_t)(a0 * c - a1 * s);  kbr[i1] = (bf16_t)(a1 * c + a0 * s);
    a0 = Ki[i0]; a1 = Ki[i1];
    kbi[i0] = (bf16_t)(a0 * c - a1 * s);  kbi[i1] = (bf16_t)(a1 * c + a0 * s);
}

// ---------------- V transpose: [B,L,H*HD] fp32 -> [B,H,HD,L] bf16 ----------------
__global__ void vtrans_kernel(const float* __restrict__ Vr, const float* __restrict__ Vi,
                              bf16_t* __restrict__ Vtr, bf16_t* __restrict__ Vti) {
    int t = blockIdx.x * blockDim.x + threadIdx.x;   // B*H*HD*L = NB
    if (t >= (int)NB) return;
    int l = t & (Lc - 1);
    int d = (t >> 10) & (HDc - 1);
    int h = (t >> 16) & (Hc - 1);
    int b = t >> 20;
    size_t src = ((size_t)(b * Lc + l)) * Dc + h * HDc + d;
    Vtr[t] = (bf16_t)Vr[src];
    Vti[t] = (bf16_t)Vi[src];
}

// ---------------- Flash attention: one wave per 16-query tile ----------------
// scores = Qr.Kr + Qi.Ki (Hermitian real part), causal softmax, O = P @ V (r,i).
#define ATT_WAVES 8
__global__ __launch_bounds__(256, 1)
void attn_kernel(const bf16_t* __restrict__ qbr, const bf16_t* __restrict__ qbi,
                 const bf16_t* __restrict__ kbr, const bf16_t* __restrict__ kbi,
                 const bf16_t* __restrict__ vtr, const bf16_t* __restrict__ vti,
                 float* __restrict__ Or, float* __restrict__ Oi) {
    alignas(16) __shared__ bf16_t pbuf[ATT_WAVES][16 * 32];
    int lane = threadIdx.x & 31;
    int wv   = threadIdx.x >> 5;
    int wid  = blockIdx.x * ATT_WAVES + wv;       // B*H*(L/16) = 2048
    int qt = wid & 63;
    int h  = (wid >> 6) & (Hc - 1);
    int b  = wid >> 10;
    int qbase = qt * 16;
    const size_t hoff = (size_t)h * HDc;
    const bf16_t* qrb = qbr + ((size_t)(b * Lc + qbase)) * Dc + hoff;
    const bf16_t* qib = qbi + ((size_t)(b * Lc + qbase)) * Dc + hoff;
    v16bf fqr0 = load_frag_a(qrb, Dc, lane);
    v16bf fqr1 = load_frag_a(qrb + 32, Dc, lane);
    v16bf fqi0 = load_frag_a(qib, Dc, lane);
    v16bf fqi1 = load_frag_a(qib + 32, Dc, lane);

    float mst[8], lst[8];
    v8f zero8 = {0,0,0,0,0,0,0,0};
    v8f o_r[4], o_i[4];
#pragma unroll
    for (int j = 0; j < 4; ++j) { o_r[j] = zero8; o_i[j] = zero8; }
#pragma unroll
    for (int j = 0; j < 8; ++j) { mst[j] = -1e30f; lst[j] = 0.f; }
    int col  = lane & 15;
    int rofs = (lane >> 4) ? 8 : 0;
    bf16_t* pb = pbuf[wv];

    for (int kb = 0; kb <= qbase + 15; kb += 32) {
        const bf16_t* kr0 = kbr + ((size_t)(b * Lc + kb)) * Dc + hoff;
        const bf16_t* ki0 = kbi + ((size_t)(b * Lc + kb)) * Dc + hoff;
        const bf16_t* kr1 = kr0 + (size_t)16 * Dc;
        const bf16_t* ki1 = ki0 + (size_t)16 * Dc;
        v8f s0 = zero8, s1 = zero8;
        s0 = wmma_bf16(fqr0, load_frag_b(kr0, Dc, lane), s0);
        s0 = wmma_bf16(fqr1, load_frag_b(kr0 + 32, Dc, lane), s0);
        s0 = wmma_bf16(fqi0, load_frag_b(ki0, Dc, lane), s0);
        s0 = wmma_bf16(fqi1, load_frag_b(ki0 + 32, Dc, lane), s0);
        s1 = wmma_bf16(fqr0, load_frag_b(kr1, Dc, lane), s1);
        s1 = wmma_bf16(fqr1, load_frag_b(kr1 + 32, Dc, lane), s1);
        s1 = wmma_bf16(fqi0, load_frag_b(ki1, Dc, lane), s1);
        s1 = wmma_bf16(fqi1, load_frag_b(ki1 + 32, Dc, lane), s1);
        // scale + causal mask + online softmax (row = 16 lanes within half-wave)
        float alpha[8];
#pragma unroll
        for (int j = 0; j < 8; ++j) {
            int q = qbase + rofs + j;
            float v0 = s0[j] * SCALEc;
            float v1 = s1[j] * SCALEc;
            if (kb + col > q)      v0 = -1e30f;
            if (kb + 16 + col > q) v1 = -1e30f;
            float t = fmaxf(v0, v1);
            for (int m = 1; m < 16; m <<= 1) t = fmaxf(t, __shfl_xor(t, m, 32));
            float mnew = fmaxf(mst[j], t);
            float a  = __expf(mst[j] - mnew);
            float p0 = __expf(v0 - mnew);
            float p1 = __expf(v1 - mnew);
            float rs = p0 + p1;
            for (int m = 1; m < 16; m <<= 1) rs += __shfl_xor(rs, m, 32);
            lst[j] = lst[j] * a + rs;
            mst[j] = mnew;
            alpha[j] = a;
            s0[j] = p0; s1[j] = p1;
        }
#pragma unroll
        for (int ht = 0; ht < 4; ++ht)
#pragma unroll
            for (int j = 0; j < 8; ++j) { o_r[ht][j] *= alpha[j]; o_i[ht][j] *= alpha[j]; }
        // stage P (16x32) through LDS to re-swizzle into an A fragment
#pragma unroll
        for (int j = 0; j < 8; ++j) {
            int r = rofs + j;
            pb[r * 32 + col]      = (bf16_t)s0[j];
            pb[r * 32 + 16 + col] = (bf16_t)s1[j];
        }
        asm volatile("s_wait_dscnt 0x0" ::: "memory");
        v16bf fp = load_frag_a(pb, 32, lane);
#pragma unroll
        for (int ht = 0; ht < 4; ++ht) {
            const bf16_t* vr = vtr + (((size_t)(b * Hc + h)) * HDc + ht * 16) * Lc + kb;
            const bf16_t* vi = vti + (((size_t)(b * Hc + h)) * HDc + ht * 16) * Lc + kb;
            o_r[ht] = wmma_bf16(fp, load_frag_b(vr, Lc, lane), o_r[ht]);
            o_i[ht] = wmma_bf16(fp, load_frag_b(vi, Lc, lane), o_i[ht]);
        }
    }
    // normalize + store merged-head [B,L,D]
#pragma unroll
    for (int j = 0; j < 8; ++j) {
        int q = qbase + rofs + j;
        float inv = 1.0f / lst[j];
        size_t base = ((size_t)(b * Lc + q)) * Dc + hoff + col;
#pragma unroll
        for (int ht = 0; ht < 4; ++ht) {
            Or[base + ht * 16] = o_r[ht][j] * inv;
            Oi[base + ht * 16] = o_i[ht][j] * inv;
        }
    }
}

// ---------------- host-side orchestration ----------------
extern "C" void kernel_launch(void* const* d_in, const int* in_sizes, int n_in,
                              void* d_out, int out_size, void* d_ws, size_t ws_size,
                              hipStream_t stream) {
    const float* x_r   = (const float*)d_in[0];
    const float* x_i   = (const float*)d_in[1];
    const float* Wq_r  = (const float*)d_in[2];
    const float* Wq_i  = (const float*)d_in[3];
    const float* Wk_r  = (const float*)d_in[4];
    const float* Wk_i  = (const float*)d_in[5];
    const float* Wv_r  = (const float*)d_in[6];
    const float* Wv_i  = (const float*)d_in[7];
    const float* Wo_r  = (const float*)d_in[8];
    const float* Wo_i  = (const float*)d_in[9];
    const float* bo_r  = (const float*)d_in[10];
    const float* bo_i  = (const float*)d_in[11];
    const float* ln1gr = (const float*)d_in[12];
    const float* ln1gi = (const float*)d_in[13];
    const float* ln1br = (const float*)d_in[14];
    const float* ln1bi = (const float*)d_in[15];
    const float* ln2gr = (const float*)d_in[16];
    const float* ln2gi = (const float*)d_in[17];
    const float* ln2br = (const float*)d_in[18];
    const float* ln2bi = (const float*)d_in[19];
    const float* W1_r  = (const float*)d_in[20];
    const float* W1_i  = (const float*)d_in[21];
    const float* b1_r  = (const float*)d_in[22];
    const float* b1_i  = (const float*)d_in[23];
    const float* W2_r  = (const float*)d_in[24];
    const float* W2_i  = (const float*)d_in[25];
    const float* b2_r  = (const float*)d_in[26];
    const float* b2_i  = (const float*)d_in[27];
    const float* mod_b = (const float*)d_in[28];

    char* ws = (char*)d_ws;
    size_t off = 0;
    auto alloc = [&](size_t bytes) -> char* {
        char* p = ws + off;
        off += (bytes + 255) & ~(size_t)255;
        return p;
    };
    // fp32 activation buffers
    float* bQr = (float*)alloc(NB * 4);
    float* bQi = (float*)alloc(NB * 4);
    float* bKr = (float*)alloc(NB * 4);
    float* bKi = (float*)alloc(NB * 4);
    float* bVr = (float*)alloc(NB * 4);
    float* bVi = (float*)alloc(NB * 4);
    float* bar = (float*)alloc(NB * 4);
    float* bai = (float*)alloc(NB * 4);
    // bf16 activation buffers
    bf16_t* h1r = (bf16_t*)alloc(NB * 2);      // ln1 out; reused as O bf16
    bf16_t* h1i = (bf16_t*)alloc(NB * 2);
    bf16_t* qbr = (bf16_t*)alloc(NB * 2);
    bf16_t* qbi = (bf16_t*)alloc(NB * 2);
    bf16_t* kbr = (bf16_t*)alloc(NB * 2);
    bf16_t* kbi = (bf16_t*)alloc(NB * 2);
    bf16_t* vtr = (bf16_t*)alloc(NB * 2);
    bf16_t* vti = (bf16_t*)alloc(NB * 2);
    bf16_t* h2r = (bf16_t*)alloc(NB * 2);
    bf16_t* h2i = (bf16_t*)alloc(NB * 2);
    bf16_t* fbr = (bf16_t*)alloc(NHID * 2);
    bf16_t* fbi = (bf16_t*)alloc(NHID * 2);
    // bf16 weights
    bf16_t* wqr = (bf16_t*)alloc((size_t)Dc * Dc * 2);
    bf16_t* wqi = (bf16_t*)alloc((size_t)Dc * Dc * 2);
    bf16_t* wkr = (bf16_t*)alloc((size_t)Dc * Dc * 2);
    bf16_t* wki = (bf16_t*)alloc((size_t)Dc * Dc * 2);
    bf16_t* wvr = (bf16_t*)alloc((size_t)Dc * Dc * 2);
    bf16_t* wvi = (bf16_t*)alloc((size_t)Dc * Dc * 2);
    bf16_t* wor = (bf16_t*)alloc((size_t)Dc * Dc * 2);
    bf16_t* woi = (bf16_t*)alloc((size_t)Dc * Dc * 2);
    bf16_t* w1r = (bf16_t*)alloc((size_t)Dc * HIDc * 2);
    bf16_t* w1i = (bf16_t*)alloc((size_t)Dc * HIDc * 2);
    bf16_t* w2r = (bf16_t*)alloc((size_t)Dc * HIDc * 2);
    bf16_t* w2i = (bf16_t*)alloc((size_t)Dc * HIDc * 2);

    auto conv = [&](const float* s, bf16_t* d, size_t n) {
        f2b_kernel<<<(int)((n + 255) / 256), 256, 0, stream>>>(s, d, (int)n);
    };
    auto gemm = [&](const bf16_t* Xr, const bf16_t* Xi, const bf16_t* Wr, const bf16_t* Wi,
                    float* Yr, float* Yi, bf16_t* Ybr, bf16_t* Ybi,
                    const float* br, const float* bi, const float* Rr, const float* Ri,
                    const float* mb, int M, int N, int K, int mode) {
        int waves = (M / 32) * (N / 32);   // 32x32 wave tiles
        cgemm_kernel<<<waves / 8, 256, 0, stream>>>(Xr, Xi, Wr, Wi, Yr, Yi, Ybr, Ybi,
                                                    br, bi, Rr, Ri, mb, M, N, K, mode);
    };

    // 1) weights -> bf16
    conv(Wq_r, wqr, (size_t)Dc * Dc);  conv(Wq_i, wqi, (size_t)Dc * Dc);
    conv(Wk_r, wkr, (size_t)Dc * Dc);  conv(Wk_i, wki, (size_t)Dc * Dc);
    conv(Wv_r, wvr, (size_t)Dc * Dc);  conv(Wv_i, wvi, (size_t)Dc * Dc);
    conv(Wo_r, wor, (size_t)Dc * Dc);  conv(Wo_i, woi, (size_t)Dc * Dc);
    conv(W1_r, w1r, (size_t)Dc * HIDc); conv(W1_i, w1i, (size_t)Dc * HIDc);
    conv(W2_r, w2r, (size_t)Dc * HIDc); conv(W2_i, w2i, (size_t)Dc * HIDc);

    // 2) LN1
    cln_kernel<<<MR, 256, 0, stream>>>(x_r, x_i, ln1gr, ln1gi, ln1br, ln1bi, h1r, h1i);
    // 3-5) Q, K, V projections (fp32 out)
    gemm(h1r, h1i, wqr, wqi, bQr, bQi, nullptr, nullptr, nullptr, nullptr, nullptr, nullptr, nullptr, MR, Dc, Dc, 0);
    gemm(h1r, h1i, wkr, wki, bKr, bKi, nullptr, nullptr, nullptr, nullptr, nullptr, nullptr, nullptr, MR, Dc, Dc, 0);
    gemm(h1r, h1i, wvr, wvi, bVr, bVi, nullptr, nullptr, nullptr, nullptr, nullptr, nullptr, nullptr, MR, Dc, Dc, 0);
    // 6) RoPE -> bf16 Q,K
    rope_kernel<<<(Bc * Lc * Hc * 32) / 256, 256, 0, stream>>>(bQr, bQi, bKr, bKi, qbr, qbi, kbr, kbi);
    // 7) V transpose -> bf16 [B,H,HD,L]
    vtrans_kernel<<<(int)(NB / 256), 256, 0, stream>>>(bVr, bVi, vtr, vti);
    // 8) attention (reuse bQr/bQi as O fp32)
    attn_kernel<<<(Bc * Hc * (Lc / 16)) / ATT_WAVES, ATT_WAVES * 32, 0, stream>>>(
        qbr, qbi, kbr, kbi, vtr, vti, bQr, bQi);
    // 9) O -> bf16 (reuse h1)
    conv(bQr, h1r, NB);  conv(bQi, h1i, NB);
    // 10) output projection + bias + residual(x) -> ar/ai
    gemm(h1r, h1i, wor, woi, bar, bai, nullptr, nullptr, bo_r, bo_i, x_r, x_i, nullptr, MR, Dc, Dc, 0);
    // 11) LN2
    cln_kernel<<<MR, 256, 0, stream>>>(bar, bai, ln2gr, ln2gi, ln2br, ln2bi, h2r, h2i);
    // 12) FFN1 + bias + ModReLU fused -> bf16 hidden
    gemm(h2r, h2i, w1r, w1i, nullptr, nullptr, fbr, fbi, b1_r, b1_i, nullptr, nullptr, mod_b, MR, HIDc, Dc, 1);
    // 13) FFN2 + bias + residual(ar/ai) -> d_out (r then i)
    float* out_r = (float*)d_out;
    float* out_i = out_r + NB;
    gemm(fbr, fbi, w2r, w2i, out_r, out_i, nullptr, nullptr, b2_r, b2_i, bar, bai, nullptr, MR, Dc, HIDc, 0);
}
